// ShareDeLayer_43611097924203
// MI455X (gfx1250) — matrix operand order repaired and checked by reading, not compiled, not run on hardware
//
#include <hip/hip_runtime.h>

typedef _Float16 h16;
typedef __attribute__((ext_vector_type(16))) _Float16 v16h;
typedef __attribute__((ext_vector_type(8)))  _Float16 v8h;
typedef __attribute__((ext_vector_type(4)))  _Float16 v4h;
typedef __attribute__((ext_vector_type(8)))  float    v8f;

constexpr int Q_  = 1024;
constexpr int B_  = 8;
constexpr int D_  = 256;
constexpr int H_  = 8;
constexpr int S_  = 1024;
constexpr int FF_ = 1024;
constexpr int E_  = 8;
constexpr int T_  = Q_ * B_;   // 8192 tokens
constexpr int HD  = D_ / H_;   // 32 head dim
constexpr int CAP = 3072;      // per-expert token capacity (E[count]=2048, sigma~39)

// ---------------------------------------------------------------------------
// WMMA helpers (CDNA5: D = A[16x32 f16] x B[32x16 f16] + C[16x16 f32])
// ---------------------------------------------------------------------------
__device__ __forceinline__ v8f wmma32(v16h a, v16h b, v8f c) {
  return __builtin_amdgcn_wmma_f32_16x16x32_f16(false, a, false, b, (short)0, c,
                                                false, false);
}

// A fragment: lane L -> row m=L&15; halfs j -> K = (L>>4)*16 + j
//             (two 16B loads)
__device__ __forceinline__ v16h load_a_frag(const h16* __restrict__ A, size_t lda,
                                            int row0, int k0, int lane) {
  int m = lane & 15, lh = lane >> 4;
  const h16* p = A + (size_t)(row0 + m) * lda + (size_t)(k0 + lh * 16);
  union { v16h v; v8h q[2]; } u;
  u.q[0] = *(const v8h*)p;
  u.q[1] = *(const v8h*)(p + 8);
  return u.v;
}

__device__ __forceinline__ v16h load_a_frag_gather(const h16* __restrict__ A, size_t lda,
                                                   const int* __restrict__ idx, int count,
                                                   int row0, int k0, int lane) {
  int m = lane & 15, lh = lane >> 4;
  int r = row0 + m;
  if (r >= count) r = count - 1;
  const h16* p = A + (size_t)idx[r] * lda + (size_t)(k0 + lh * 16);
  union { v16h v; v8h q[2]; } u;
  u.q[0] = *(const v8h*)p;
  u.q[1] = *(const v8h*)(p + 8);
  return u.v;
}

// B fragment (B = W^T, W row-major [N,K]): lane L -> col n=L&15;
// halfs j -> K = (L>>4)*16 + j  (one contiguous 32B load from W row n)
__device__ __forceinline__ v16h load_b_frag(const h16* __restrict__ W, size_t ldw,
                                            int n0, int k0, int lane) {
  int n = lane & 15, lh = lane >> 4;
  return *(const v16h*)(W + (size_t)(n0 + n) * ldw + (size_t)(k0 + lh * 16));
}

// ---------------------------------------------------------------------------
// Generic GEMM: C[M,N] = A[M,K] * W[N,K]^T (+bias) (+residual)
// block: 256 thr = 8 waves (4 M-waves x 2 N-waves); block tile 128x64,
// wave tile 32x32 (2x2 WMMA tiles), K-step 32; K is compile time -> unrolled.
// MODE 0: f16 out = c + bias           (Q/K projections)
// MODE 1: f32 out = c + bias + residual[(row/resDiv)*ldc + col]
// MODE 2: f16 V^T scatter: vpT[(row&7)*D + col][row>>3]  (fused V transpose)
// ---------------------------------------------------------------------------
template <int MODE, int K>
__global__ void k_gemm(const h16* __restrict__ A, int lda,
                       const h16* __restrict__ W, int ldw,
                       const float* __restrict__ bias,
                       void* __restrict__ Cout, int ldc,
                       const float* __restrict__ residual, int resDiv) {
  int lane = threadIdx.x & 31, wave = threadIdx.x >> 5;
  int rowBase = blockIdx.x * 128 + (wave >> 1) * 32;
  int colBase = blockIdx.y * 64 + (wave & 1) * 32;

  const v8f z = {0.f, 0.f, 0.f, 0.f, 0.f, 0.f, 0.f, 0.f};
  v8f acc[2][2] = {{z, z}, {z, z}};

#pragma unroll
  for (int k0 = 0; k0 < K; k0 += 32) {
    v16h a0 = load_a_frag(A, lda, rowBase, k0, lane);
    v16h a1 = load_a_frag(A, lda, rowBase + 16, k0, lane);
    v16h b0 = load_b_frag(W, ldw, colBase, k0, lane);
    v16h b1 = load_b_frag(W, ldw, colBase + 16, k0, lane);
    acc[0][0] = wmma32(a0, b0, acc[0][0]);
    acc[0][1] = wmma32(a0, b1, acc[0][1]);
    acc[1][0] = wmma32(a1, b0, acc[1][0]);
    acc[1][1] = wmma32(a1, b1, acc[1][1]);
  }

  int lh = lane >> 4, ln = lane & 15;
#pragma unroll
  for (int ti = 0; ti < 2; ++ti)
#pragma unroll
    for (int tj = 0; tj < 2; ++tj)
#pragma unroll
      for (int r = 0; r < 8; ++r) {
        int row = rowBase + ti * 16 + lh * 8 + r;
        int col = colBase + tj * 16 + ln;
        float v = acc[ti][tj][r] + bias[col];
        if constexpr (MODE == 0) {
          ((h16*)Cout)[(size_t)row * ldc + col] = (h16)v;
        } else if constexpr (MODE == 1) {
          ((float*)Cout)[(size_t)row * ldc + col] =
              v + residual[(size_t)(row / resDiv) * ldc + col];
        } else {  // MODE 2: fused transpose store for V: row = s*B+b
          ((h16*)Cout)[(size_t)((row & 7) * D_ + col) * S_ + (row >> 3)] = (h16)v;
        }
      }
}

// ---------------------------------------------------------------------------
// Flash attention, transposed-score form. One (b,h) per blockIdx.y;
// 8 waves x 16 q-columns per block; 32 keys per iteration.
//   S^T = K(16x32d) x Q^T(32d x 16q)  -> lane owns one q column, 16 scores
//   softmax: per-lane VALU reduction + single shfl_xor(16)
//   P^T already matches the A-fragment element ownership after one half-swap
//   O^T = V^T(16d x 32keys) x P(32keys x 16q), rescale = per-lane scalar mult
// kp layout: [s*B+b, D] f16 ; vpT layout: [(b*H+h)*32+d, S] f16.
// ---------------------------------------------------------------------------
__global__ void k_attention(const h16* __restrict__ qp, int qMulQ, int qMulB,
                            const h16* __restrict__ kp,
                            const h16* __restrict__ vpT,
                            h16* __restrict__ attnO) {
  int lane = threadIdx.x & 31, wave = threadIdx.x >> 5;
  int bh = blockIdx.y;
  int b = bh >> 3, h = bh & 7;
  int q0 = blockIdx.x * 128 + wave * 16;
  int ln = lane & 15, lh = lane >> 4;

  // Q^T B-fragment: lane -> q col = ln, halfs j -> d = lh*16 + j (one 32B load)
  int qg = q0 + ln;
  v16h qb = *(const v16h*)(qp + (size_t)(qg * qMulQ + b * qMulB) * D_ + h * HD + lh * 16);

  const v8f z = {0.f, 0.f, 0.f, 0.f, 0.f, 0.f, 0.f, 0.f};
  v8f ot0 = z, ot1 = z;            // O^T tiles: d rows [0..15] / [16..31], q cols
  float mprev = -3.0e38f, lsum = 0.f;
  const float scale = 0.17677669529663687f;   // 1/sqrt(32)
  const h16* vbase = vpT + (size_t)((b * H_ + h) * HD) * S_;

  for (int kt = 0; kt < S_; kt += 32) {
    // K A-fragments: rows = keys, K-dim = d(32)
    union { v16h v; v8h q[2]; } ka0, ka1;
    const h16* kr0 = kp + (size_t)((kt + ln) * B_ + b) * D_ + h * HD;
    const h16* kr1 = kp + (size_t)((kt + 16 + ln) * B_ + b) * D_ + h * HD;
    ka0.q[0] = *(const v8h*)(kr0 + lh * 16);
    ka0.q[1] = *(const v8h*)(kr0 + lh * 16 + 8);
    ka1.q[0] = *(const v8h*)(kr1 + lh * 16);
    ka1.q[1] = *(const v8h*)(kr1 + lh * 16 + 8);

    v8f st0 = wmma32(ka0.v, qb, z);   // keys kt+lh*8+r        (x q cols)
    v8f st1 = wmma32(ka1.v, qb, z);   // keys kt+16+lh*8+r     (x q cols)

    // online softmax for this lane's q column
    float mx = -3.0e38f;
#pragma unroll
    for (int r = 0; r < 8; ++r) mx = fmaxf(mx, fmaxf(st0[r], st1[r]));
    mx *= scale;
    mx = fmaxf(mx, __shfl_xor(mx, 16, 32));
    float mnew = fmaxf(mprev, mx);
    float alpha = __expf(mprev - mnew);
    mprev = mnew;

    v8h pk0, pk1;
    float rs = 0.f;
#pragma unroll
    for (int r = 0; r < 8; ++r) {
      float p0 = __expf(st0[r] * scale - mnew);
      float p1 = __expf(st1[r] * scale - mnew);
      rs += p0 + p1;
      pk0[r] = (h16)p0;
      pk1[r] = (h16)p1;
    }
    rs += __shfl_xor(rs, 16, 32);
    lsum = lsum * alpha + rs;
    ot0 *= alpha;
    ot1 *= alpha;

    // Build P B-fragment (halfs j: key = lh*16 + j) via one half-swap
    union { v8h h8; int i[4]; } a0u, a1u, x0u, x1u;
    a0u.h8 = pk0; a1u.h8 = pk1;
#pragma unroll
    for (int i = 0; i < 4; ++i) {
      x0u.i[i] = __shfl_xor(a0u.i[i], 16, 32);
      x1u.i[i] = __shfl_xor(a1u.i[i], 16, 32);
    }
    union { v16h v; v8h q[2]; } pb;
    pb.q[0] = lh ? x1u.h8 : a0u.h8;   // keys lh*16 + [0..7]
    pb.q[1] = lh ? a1u.h8 : x0u.h8;   // keys lh*16 + [8..15]

    // V^T A-fragments (rows = d, K-dim = 32 keys; contiguous in vpT)
    union { v16h v; v8h q[2]; } va0, va1;
    const h16* vr0 = vbase + (size_t)ln * S_ + kt;
    const h16* vr1 = vbase + (size_t)(16 + ln) * S_ + kt;
    va0.q[0] = *(const v8h*)(vr0 + lh * 16);
    va0.q[1] = *(const v8h*)(vr0 + lh * 16 + 8);
    va1.q[0] = *(const v8h*)(vr1 + lh * 16);
    va1.q[1] = *(const v8h*)(vr1 + lh * 16 + 8);

    ot0 = wmma32(va0.v, pb.v, ot0);
    ot1 = wmma32(va1.v, pb.v, ot1);
  }

  float inv = 1.f / lsum;
  v8h s0, s1;
#pragma unroll
  for (int r = 0; r < 8; ++r) {
    s0[r] = (h16)(ot0[r] * inv);
    s1[r] = (h16)(ot1[r] * inv);
  }
  size_t orow = (size_t)(qg * B_ + b) * D_ + h * HD;  // d rows lh*8+r contiguous
  *(v8h*)(attnO + orow + lh * 8) = s0;
  *(v8h*)(attnO + orow + 16 + lh * 8) = s1;
}

// ---------------------------------------------------------------------------
// LayerNorm over D=256; one row per wave (8 floats/lane). Emits f32 + f16.
// ---------------------------------------------------------------------------
__global__ void k_layernorm(const float* __restrict__ x,
                            const float* __restrict__ g,
                            const float* __restrict__ be,
                            float* __restrict__ y32, h16* __restrict__ y16) {
  int lane = threadIdx.x & 31, wave = threadIdx.x >> 5;
  int row = blockIdx.x * 8 + wave;
  const float4* p = (const float4*)(x + (size_t)row * D_ + lane * 8);
  float4 a0 = p[0], a1 = p[1];
  float v[8] = {a0.x, a0.y, a0.z, a0.w, a1.x, a1.y, a1.z, a1.w};
  float s = 0.f, ss = 0.f;
#pragma unroll
  for (int i = 0; i < 8; ++i) { s += v[i]; ss += v[i] * v[i]; }
#pragma unroll
  for (int off = 16; off >= 1; off >>= 1) {
    s += __shfl_xor(s, off, 32);
    ss += __shfl_xor(ss, off, 32);
  }
  float mean = s * (1.f / D_);
  float var = ss * (1.f / D_) - mean * mean;
  float rstd = rsqrtf(var + 1e-5f);
  int c0 = lane * 8;
#pragma unroll
  for (int i = 0; i < 8; ++i) {
    int c = c0 + i;
    float o = (v[i] - mean) * rstd * g[c] + be[c];
    y32[(size_t)row * D_ + c] = o;
    y16[(size_t)row * D_ + c] = (h16)o;
  }
}

// ---------------------------------------------------------------------------
// MoE gate: logits = x @ gate_w^T + gate_b ; top-2 ; softmax over the 2;
// atomic-append token into per-expert compacted lists.
// ---------------------------------------------------------------------------
__global__ void k_gate(const float* __restrict__ x,
                       const float* __restrict__ gw, const float* __restrict__ gb,
                       int* __restrict__ cnt, int* __restrict__ idxList,
                       float* __restrict__ gateW) {
  __shared__ float sgw[E_ * D_];
  for (int i = threadIdx.x; i < E_ * D_; i += blockDim.x) sgw[i] = gw[i];
  __syncthreads();
  int t = blockIdx.x * blockDim.x + threadIdx.x;
  float acc[E_];
#pragma unroll
  for (int e = 0; e < E_; ++e) acc[e] = gb[e];
  for (int d = 0; d < D_; ++d) {
    float xv = x[(size_t)t * D_ + d];
#pragma unroll
    for (int e = 0; e < E_; ++e) acc[e] += xv * sgw[e * D_ + d];
  }
  int i0 = 0;
#pragma unroll
  for (int e = 1; e < E_; ++e) if (acc[e] > acc[i0]) i0 = e;
  int i1 = (i0 == 0) ? 1 : 0;
#pragma unroll
  for (int e = 0; e < E_; ++e) if (e != i0 && acc[e] > acc[i1]) i1 = e;
  float w0 = 1.f / (1.f + __expf(acc[i1] - acc[i0]));
  float w1 = 1.f - w0;
  int s0 = atomicAdd(&cnt[i0], 1);
  if (s0 < CAP) { idxList[i0 * CAP + s0] = t; gateW[i0 * CAP + s0] = w0; }
  int s1 = atomicAdd(&cnt[i1], 1);
  if (s1 < CAP) { idxList[i1 * CAP + s1] = t; gateW[i1 * CAP + s1] = w1; }
}

// ---------------------------------------------------------------------------
// MoE expert GEMM 1: h = relu(gather(X) @ W1[e]^T + b1[e]) -> compact hBuf
// ---------------------------------------------------------------------------
__global__ void k_moe_ffn1(const h16* __restrict__ X,
                           const h16* __restrict__ W1, const float* __restrict__ B1,
                           const int* __restrict__ idxAll, const int* __restrict__ cnt,
                           h16* __restrict__ hBuf) {
  int e = blockIdx.z;
  int count = cnt[e]; if (count > CAP) count = CAP;
  if ((int)blockIdx.x * 128 >= count) return;
  int lane = threadIdx.x & 31, wave = threadIdx.x >> 5;
  int rowBase = blockIdx.x * 128 + (wave >> 1) * 32;
  int colBase = blockIdx.y * 64 + (wave & 1) * 32;
  const int* idx = idxAll + e * CAP;
  const h16* W = W1 + (size_t)e * FF_ * D_;
  const float* bias = B1 + (size_t)e * FF_;
  h16* hOut = hBuf + (size_t)e * CAP * FF_;

  const v8f z = {0.f, 0.f, 0.f, 0.f, 0.f, 0.f, 0.f, 0.f};
  v8f acc[2][2] = {{z, z}, {z, z}};
#pragma unroll
  for (int k0 = 0; k0 < D_; k0 += 32) {
    v16h a0 = load_a_frag_gather(X, D_, idx, count, rowBase, k0, lane);
    v16h a1 = load_a_frag_gather(X, D_, idx, count, rowBase + 16, k0, lane);
    v16h b0 = load_b_frag(W, D_, colBase, k0, lane);
    v16h b1 = load_b_frag(W, D_, colBase + 16, k0, lane);
    acc[0][0] = wmma32(a0, b0, acc[0][0]);
    acc[0][1] = wmma32(a0, b1, acc[0][1]);
    acc[1][0] = wmma32(a1, b0, acc[1][0]);
    acc[1][1] = wmma32(a1, b1, acc[1][1]);
  }
  int lh = lane >> 4, ln = lane & 15;
#pragma unroll
  for (int ti = 0; ti < 2; ++ti)
#pragma unroll
    for (int tj = 0; tj < 2; ++tj)
#pragma unroll
      for (int r = 0; r < 8; ++r) {
        int row = rowBase + ti * 16 + lh * 8 + r;
        if (row >= count) continue;
        int col = colBase + tj * 16 + ln;
        float v = acc[ti][tj][r] + bias[col];
        hOut[(size_t)row * FF_ + col] = (h16)fmaxf(v, 0.f);
      }
}

// ---------------------------------------------------------------------------
// MoE expert GEMM 2: y = (h @ W2[e]^T + b2[e]) * gate ; atomic scatter-add
// ---------------------------------------------------------------------------
__global__ void k_moe_ffn2(const h16* __restrict__ hBuf,
                           const h16* __restrict__ W2, const float* __restrict__ B2,
                           const int* __restrict__ idxAll, const float* __restrict__ gwAll,
                           const int* __restrict__ cnt, float* __restrict__ out) {
  int e = blockIdx.z;
  int count = cnt[e]; if (count > CAP) count = CAP;
  if ((int)blockIdx.x * 128 >= count) return;
  int lane = threadIdx.x & 31, wave = threadIdx.x >> 5;
  int rowBase = blockIdx.x * 128 + (wave >> 1) * 32;
  int colBase = blockIdx.y * 64 + (wave & 1) * 32;
  const h16* A = hBuf + (size_t)e * CAP * FF_;
  const h16* W = W2 + (size_t)e * D_ * FF_;
  const float* bias = B2 + (size_t)e * D_;
  const int* idx = idxAll + e * CAP;
  const float* gws = gwAll + e * CAP;

  const v8f z = {0.f, 0.f, 0.f, 0.f, 0.f, 0.f, 0.f, 0.f};
  v8f acc[2][2] = {{z, z}, {z, z}};
#pragma unroll 4
  for (int k0 = 0; k0 < FF_; k0 += 32) {
    v16h a0 = load_a_frag(A, FF_, rowBase, k0, lane);
    v16h a1 = load_a_frag(A, FF_, rowBase + 16, k0, lane);
    v16h b0 = load_b_frag(W, FF_, colBase, k0, lane);
    v16h b1 = load_b_frag(W, FF_, colBase + 16, k0, lane);
    acc[0][0] = wmma32(a0, b0, acc[0][0]);
    acc[0][1] = wmma32(a0, b1, acc[0][1]);
    acc[1][0] = wmma32(a1, b0, acc[1][0]);
    acc[1][1] = wmma32(a1, b1, acc[1][1]);
  }
  int lh = lane >> 4, ln = lane & 15;
#pragma unroll
  for (int ti = 0; ti < 2; ++ti)
#pragma unroll
    for (int tj = 0; tj < 2; ++tj)
#pragma unroll
      for (int r = 0; r < 8; ++r) {
        int row = rowBase + ti * 16 + lh * 8 + r;
        if (row >= count) continue;
        int col = colBase + tj * 16 + ln;
        float v = (acc[ti][tj][r] + bias[col]) * gws[row];
        atomicAdd(&out[(size_t)idx[row] * D_ + col], v);
      }
}

// ---------------------------------------------------------------------------
// Small utility kernels
// ---------------------------------------------------------------------------
__global__ void k_f32_to_f16(const float* __restrict__ src, h16* __restrict__ dst, int n4) {
  int i = blockIdx.x * blockDim.x + threadIdx.x;
  if (i >= n4) return;
  float4 v = ((const float4*)src)[i];
  v4h o;
  o[0] = (h16)v.x; o[1] = (h16)v.y; o[2] = (h16)v.z; o[3] = (h16)v.w;
  ((v4h*)dst)[i] = o;
}

__global__ void k_zero_i32(int* __restrict__ p, int n) {
  int i = blockIdx.x * blockDim.x + threadIdx.x;
  if (i < n) p[i] = 0;
}

// ---------------------------------------------------------------------------
// Host orchestration
// ---------------------------------------------------------------------------
extern "C" void kernel_launch(void* const* d_in, const int* in_sizes, int n_in,
                              void* d_out, int out_size, void* d_ws, size_t ws_size,
                              hipStream_t stream) {
  const float* f_out   = (const float*)d_in[0];
  const float* f_mem   = (const float*)d_in[1];
  const float* f_embed = (const float*)d_in[2];
  const float* sa_wi = (const float*)d_in[3];
  const float* sa_bi = (const float*)d_in[4];
  const float* sa_wo = (const float*)d_in[5];
  const float* sa_bo = (const float*)d_in[6];
  const float* ca_wi = (const float*)d_in[7];
  const float* ca_bi = (const float*)d_in[8];
  const float* ca_wo = (const float*)d_in[9];
  const float* ca_bo = (const float*)d_in[10];
  const float* n1_g = (const float*)d_in[11];
  const float* n1_b = (const float*)d_in[12];
  const float* n2_g = (const float*)d_in[13];
  const float* n2_b = (const float*)d_in[14];
  const float* gate_w = (const float*)d_in[15];
  const float* gate_b = (const float*)d_in[16];
  const float* w1 = (const float*)d_in[17];
  const float* b1 = (const float*)d_in[18];
  const float* w2 = (const float*)d_in[19];
  const float* b2 = (const float*)d_in[20];
  float* outP = (float*)d_out;

  char* ws = (char*)d_ws;
  size_t off = 0;
  auto carve = [&](size_t bytes) -> void* {
    void* p = ws + off;
    off += (bytes + 255) & ~(size_t)255;
    return p;
  };

  h16* embed16 = (h16*)carve((size_t)Q_ * D_ * 2);
  h16* mem16   = (h16*)carve((size_t)T_ * D_ * 2);   // later reused as qp2
  h16* out16   = (h16*)carve((size_t)T_ * D_ * 2);   // later reused as tgt2f16
  h16* wsa_i   = (h16*)carve((size_t)3 * D_ * D_ * 2);
  h16* wsa_o   = (h16*)carve((size_t)D_ * D_ * 2);
  h16* wca_i   = (h16*)carve((size_t)3 * D_ * D_ * 2);
  h16* wca_o   = (h16*)carve((size_t)D_ * D_ * 2);
  h16* w1h     = (h16*)carve((size_t)E_ * FF_ * D_ * 2);
  h16* w2h     = (h16*)carve((size_t)E_ * D_ * FF_ * 2);
  h16* qp1     = (h16*)carve((size_t)Q_ * D_ * 2);
  h16* kp      = (h16*)carve((size_t)T_ * D_ * 2);
  h16* vpT     = (h16*)carve((size_t)T_ * D_ * 2);
  h16* attnO   = (h16*)carve((size_t)T_ * D_ * 2);
  float* resbuf  = (float*)carve((size_t)T_ * D_ * 4);
  float* tgt1f32 = (float*)carve((size_t)T_ * D_ * 4);
  h16* tgt1f16   = (h16*)carve((size_t)T_ * D_ * 2);
  int* cnt       = (int*)carve((size_t)E_ * 4);
  int* idxList   = (int*)carve((size_t)E_ * CAP * 4);
  float* gateW   = (float*)carve((size_t)E_ * CAP * 4);
  h16* hBuf      = (h16*)carve((size_t)E_ * CAP * FF_ * 2);
  (void)ws_size; (void)in_sizes; (void)n_in; (void)out_size;

  auto conv = [&](const float* s, h16* d, int n) {
    int n4 = n / 4;
    k_f32_to_f16<<<dim3((n4 + 255) / 256), dim3(256), 0, stream>>>(s, d, n4);
  };
  conv(f_embed, embed16, Q_ * D_);
  conv(f_mem, mem16, T_ * D_);
  conv(f_out, out16, T_ * D_);
  conv(sa_wi, wsa_i, 3 * D_ * D_);
  conv(sa_wo, wsa_o, D_ * D_);
  conv(ca_wi, wca_i, 3 * D_ * D_);
  conv(ca_wo, wca_o, D_ * D_);
  conv(w1, w1h, E_ * FF_ * D_);
  conv(w2, w2h, E_ * D_ * FF_);
  k_zero_i32<<<dim3(1), dim3(32), 0, stream>>>(cnt, E_);

  dim3 blk(256);
  // -------- stage 1: cross-attn(tgt0=bcast(embed), memory) --------
  k_gemm<0, D_><<<dim3(Q_ / 128, D_ / 64), blk, 0, stream>>>(
      embed16, D_, wsa_i, D_, sa_bi, qp1, D_, nullptr, 1);
  k_gemm<0, D_><<<dim3(T_ / 128, D_ / 64), blk, 0, stream>>>(
      mem16, D_, wsa_i + (size_t)D_ * D_, D_, sa_bi + D_, kp, D_, nullptr, 1);
  k_gemm<2, D_><<<dim3(T_ / 128, D_ / 64), blk, 0, stream>>>(
      mem16, D_, wsa_i + (size_t)2 * D_ * D_, D_, sa_bi + 2 * D_, vpT, D_, nullptr, 1);
  k_attention<<<dim3(Q_ / 128, B_ * H_), blk, 0, stream>>>(qp1, 1, 0, kp, vpT, attnO);
  // out-proj + residual(tgt0[q,b] = embed[q])
  k_gemm<1, D_><<<dim3(T_ / 128, D_ / 64), blk, 0, stream>>>(
      attnO, D_, wsa_o, D_, sa_bo, resbuf, D_, f_embed, B_);
  k_layernorm<<<dim3(T_ / 8), blk, 0, stream>>>(resbuf, n1_g, n1_b, tgt1f32, tgt1f16);

  // -------- stage 2: cross-attn(tgt1, out) --------
  h16* qp2 = mem16;   // mem16 no longer needed
  k_gemm<0, D_><<<dim3(T_ / 128, D_ / 64), blk, 0, stream>>>(
      tgt1f16, D_, wca_i, D_, ca_bi, qp2, D_, nullptr, 1);
  k_gemm<0, D_><<<dim3(T_ / 128, D_ / 64), blk, 0, stream>>>(
      out16, D_, wca_i + (size_t)D_ * D_, D_, ca_bi + D_, kp, D_, nullptr, 1);
  k_gemm<2, D_><<<dim3(T_ / 128, D_ / 64), blk, 0, stream>>>(
      out16, D_, wca_i + (size_t)2 * D_ * D_, D_, ca_bi + 2 * D_, vpT, D_, nullptr, 1);
  k_attention<<<dim3(Q_ / 128, B_ * H_), blk, 0, stream>>>(qp2, B_, 1, kp, vpT, attnO);
  k_gemm<1, D_><<<dim3(T_ / 128, D_ / 64), blk, 0, stream>>>(
      attnO, D_, wca_o, D_, ca_bo, resbuf, D_, tgt1f32, 1);
  h16* tgt2f16 = out16;   // out16 no longer needed
  k_layernorm<<<dim3(T_ / 8), blk, 0, stream>>>(resbuf, n2_g, n2_b, outP, tgt2f16);

  // -------- MoE (routed top-2 of 8) ; final out = tgt2 + moe(tgt2) --------
  k_gate<<<dim3(T_ / 256), blk, 0, stream>>>(outP, gate_w, gate_b, cnt, idxList, gateW);
  k_moe_ffn1<<<dim3(CAP / 128, FF_ / 64, E_), blk, 0, stream>>>(
      tgt2f16, w1h, b1, idxList, cnt, hBuf);
  k_moe_ffn2<<<dim3(CAP / 128, D_ / 64, E_), blk, 0, stream>>>(
      hBuf, w2h, b2, idxList, gateW, cnt, outP);
}
